// RACL_Block_33071248179476
// MI455X (gfx1250) — matrix operand order repaired and checked by reading, not compiled, not run on hardware
//
#include <hip/hip_runtime.h>

// ---------------------------------------------------------------------------
// Problem constants (from reference): B=32, L=512, E=768, F=512, K=3, C=3
// ---------------------------------------------------------------------------
#define Bc 32
#define Lc 512
#define Ec 768
#define Fc 512
#define KEc (3 * Ec)   // 2304 contraction length for the convs
#define Lp (Lc + 2)    // zero-padded sequence length for SAME conv

typedef __attribute__((ext_vector_type(16))) __bf16 v16bf;
typedef __attribute__((ext_vector_type(8)))  __bf16 v8bf;
typedef __attribute__((ext_vector_type(8)))  float  v8f;

// ---------------------------------------------------------------------------
// Workspace layout (bytes). Everything is a multiple of 256.
// ---------------------------------------------------------------------------
constexpr size_t N_BLE  = (size_t)Bc * Lc * Ec;
constexpr size_t N_BLF  = (size_t)Bc * Lc * Fc;
constexpr size_t N_BLL  = (size_t)Bc * Lc * Lc;
constexpr size_t N_BLEP = (size_t)Bc * Lp * Ec;   // padded input elements
constexpr size_t N_WAF  = (size_t)KEc * Fc;
constexpr size_t N_WCE  = (size_t)KEc * Ec;

constexpr size_t OFF_XAP   = 0;                        // padded bf16 inputs
constexpr size_t OFF_XOP   = OFF_XAP   + N_BLEP * 2;
constexpr size_t OFF_XCP   = OFF_XOP   + N_BLEP * 2;
constexpr size_t OFF_Q     = OFF_XCP   + N_BLEP * 2;
constexpr size_t OFF_POST  = OFF_Q     + N_BLE * 2;    // position_att^T bf16
constexpr size_t OFF_WAT   = OFF_POST  + N_BLL * 2;
constexpr size_t OFF_WOT   = OFF_WAT   + N_WAF * 2;
constexpr size_t OFF_WCT   = OFF_WOT   + N_WAF * 2;
constexpr size_t OFF_ACONVT= OFF_WCT   + N_WCE * 2;    // aspect_conv^T  [b,F,L]
constexpr size_t OFF_OCONVT= OFF_ACONVT+ N_BLF * 2;    // opinion_conv^T [b,F,L]
constexpr size_t OFF_CCONVT= OFF_OCONVT+ N_BLF * 2;    // context_conv^T [b,E,L]
constexpr size_t OFF_AN    = OFF_CCONVT+ N_BLE * 2;    // l2norm bf16 (row major)
constexpr size_t OFF_ON    = OFF_AN    + N_BLF * 2;
constexpr size_t OFF_CN    = OFF_ON    + N_BLF * 2;
constexpr size_t OFF_ATTAO = OFF_CN    + N_BLE * 2;    // attention bf16
constexpr size_t OFF_ATTOA = OFF_ATTAO + N_BLL * 2;
constexpr size_t OFF_TMP1  = OFF_ATTOA + N_BLL * 2;    // q @ c_n^T bf16
constexpr size_t OFF_FATT  = OFF_TMP1  + N_BLL * 2;    // final word att bf16
constexpr size_t OFF_SCORE = OFF_FATT  + N_BLL * 2;    // reusable f32 [B,L,L]
constexpr size_t OFF_OC    = OFF_SCORE + N_BLL * 4;
constexpr size_t OFF_MS    = OFF_OC    + (size_t)Bc * Lc * 4;
constexpr size_t OFF_CS    = OFF_MS    + 256;

// d_out layout (flat f32, reference return order)
constexpr size_t DO_APRED = 0;
constexpr size_t DO_OPRED = DO_APRED + (size_t)Bc * Lc * 3;
constexpr size_t DO_SPRED = DO_OPRED + (size_t)Bc * Lc * 3;
constexpr size_t DO_AINT  = DO_SPRED + (size_t)Bc * Lc * 3;
constexpr size_t DO_OINT  = DO_AINT  + (size_t)Bc * Lc * 1024;
constexpr size_t DO_CINT  = DO_OINT  + (size_t)Bc * Lc * 1024;
constexpr size_t DO_CCONV = DO_CINT  + (size_t)Bc * Lc * Ec;

// ---------------------------------------------------------------------------
// WMMA fragment helpers (CDNA5 wave32 layouts, ISA 7.12.2)
// ---------------------------------------------------------------------------
// A 16x32 fragment from row-major bf16 matrix, tile origin at `A` (row0,k0)
__device__ __forceinline__ v16bf load_a_frag(const __bf16* A, int lda) {
  const int lane = threadIdx.x & 31;
  const int row  = lane & 15;
  const int half = lane >> 4;            // lanes 0-15: K 0-15; 16-31: K 16-31
  const __bf16* p = A + (size_t)row * lda + half * 16;
  v16bf a;
#pragma unroll
  for (int i = 0; i < 16; ++i) a[i] = p[i];
  return a;
}

// B 32x16 fragment where B^T is row-major (contiguous K per lane)
__device__ __forceinline__ v16bf load_b_frag_nt(const __bf16* Bt, int ldbt) {
  const int lane = threadIdx.x & 31;
  const int col  = lane & 15;
  const int half = lane >> 4;            // lanes 0-15: K 0-15; 16-31: K 16-31
  const __bf16* p = Bt + (size_t)col * ldbt + half * 16;
  v16bf b;
#pragma unroll
  for (int i = 0; i < 16; ++i) b[i] = p[i];
  return b;
}

// Software-pipelined dual-accumulator WMMA mainloop, unrolled x2 with
// explicit ping-pong stages (no register rotation copies).
// Kd must be a multiple of 64 (512 / 768 / 2304 here).
__device__ __forceinline__ void wmma_mainloop(const __bf16* Ab, int lda,
                                              const __bf16* Bb0, const __bf16* Bb1,
                                              int ldb, int Kd,
                                              v8f& acc0, v8f& acc1) {
  // stage 0 covers k, stage 1 covers k+32
  v16bf a0  = load_a_frag(Ab, lda);
  v16bf b00 = load_b_frag_nt(Bb0, ldb);
  v16bf b10 = load_b_frag_nt(Bb1, ldb);
  v16bf a1  = load_a_frag(Ab + 32, lda);
  v16bf b01 = load_b_frag_nt(Bb0 + 32, ldb);
  v16bf b11 = load_b_frag_nt(Bb1 + 32, ldb);
#pragma unroll 1
  for (int k0 = 0; k0 < Kd; k0 += 64) {
    const int kn = (k0 + 64 < Kd) ? (k0 + 64) : k0;   // clamped tail prefetch
    // compute stage 0, refill stage 0 for kn
    acc0 = __builtin_amdgcn_wmma_f32_16x16x32_bf16(false, a0, false, b00,
                                                   (short)0, acc0, false, false);
    acc1 = __builtin_amdgcn_wmma_f32_16x16x32_bf16(false, a0, false, b10,
                                                   (short)0, acc1, false, false);
    a0  = load_a_frag(Ab + kn, lda);
    b00 = load_b_frag_nt(Bb0 + kn, ldb);
    b10 = load_b_frag_nt(Bb1 + kn, ldb);
    // compute stage 1, refill stage 1 for kn+32
    acc0 = __builtin_amdgcn_wmma_f32_16x16x32_bf16(false, a1, false, b01,
                                                   (short)0, acc0, false, false);
    acc1 = __builtin_amdgcn_wmma_f32_16x16x32_bf16(false, a1, false, b11,
                                                   (short)0, acc1, false, false);
    a1  = load_a_frag(Ab + kn + 32, lda);
    b01 = load_b_frag_nt(Bb0 + kn + 32, ldb);
    b11 = load_b_frag_nt(Bb1 + kn + 32, ldb);
  }
}

// ---------------------------------------------------------------------------
// Small utility kernels
// ---------------------------------------------------------------------------
__global__ void cvt_f32_bf16(const float* __restrict__ in,
                             __bf16* __restrict__ out, size_t n) {
  size_t i = (size_t)blockIdx.x * blockDim.x + threadIdx.x;
  if (i < n) out[i] = (__bf16)in[i];
}

// f32 [B,L,E] -> bf16 [B,L+2,E] with zero rows at r=0 and r=L+1
__global__ void cvt_pad_f32_bf16(const float* __restrict__ in,
                                 __bf16* __restrict__ out) {
  size_t i = (size_t)blockIdx.x * blockDim.x + threadIdx.x;
  if (i >= N_BLEP) return;
  const int b = (int)(i / ((size_t)Lp * Ec));
  const int rem = (int)(i % ((size_t)Lp * Ec));
  const int r = rem / Ec, e = rem % Ec;
  float v = 0.f;
  if (r >= 1 && r <= Lc) v = in[((size_t)b * Lc + (r - 1)) * Ec + e];
  out[i] = (__bf16)v;
}

// in: [B,L,L] f32 -> outT[b,j,k] = in[b,k,j]  (coalesced writes)
__global__ void transpose_f32_bf16(const float* __restrict__ in,
                                   __bf16* __restrict__ outT) {
  size_t i = (size_t)blockIdx.x * blockDim.x + threadIdx.x;
  if (i >= N_BLL) return;
  const int b = (int)(i / ((size_t)Lc * Lc));
  const int rem = (int)(i % ((size_t)Lc * Lc));
  const int j = rem / Lc, k = rem % Lc;
  outT[i] = (__bf16)in[((size_t)b * Lc + k) * Lc + j];
}

// W[KE, N] row-major -> WT[N, KE] bf16
__global__ void wt_transpose(const float* __restrict__ W,
                             __bf16* __restrict__ WT, int KE, int N) {
  size_t i = (size_t)blockIdx.x * blockDim.x + threadIdx.x;
  if (i < (size_t)KE * N) {
    int ke = (int)(i / N), n = (int)(i % N);
    WT[(size_t)n * KE + ke] = (__bf16)W[i];
  }
}

// ---------------------------------------------------------------------------
// Conv1d (K=3, SAME) on zero-padded input == plain GEMM with lda=Ec, Kd=3*Ec:
//   A row m lives at xpad[b, l0+m, :] and the 2304-long contraction is
//   contiguous because adjacent rows are adjacent in memory.
// Writes f32 row-major (ldc) into d_out and bf16 TRANSPOSED [b, Nout, Lc].
// ---------------------------------------------------------------------------
__global__ __launch_bounds__(32)
void conv1d_wmma(const __bf16* __restrict__ xpad, const __bf16* __restrict__ WT,
                 const float* __restrict__ bias,
                 float* __restrict__ out_f32, int ldc,
                 __bf16* __restrict__ outT_bf16, int Nout) {
  const int ntn = Nout / 32, ntl = Lc / 16;
  const int tile = blockIdx.x;
  const int b  = tile / (ntl * ntn);
  const int rm = tile % (ntl * ntn);
  const int l0 = (rm / ntn) * 16, n0 = (rm % ntn) * 32;

  const __bf16* Ab  = xpad + ((size_t)b * Lp + l0) * Ec;  // row m -> l0+m-1 (+1 pad)
  const __bf16* Bb0 = WT + (size_t)n0 * KEc;
  const __bf16* Bb1 = Bb0 + (size_t)16 * KEc;

  v8f acc0 = {0.f, 0.f, 0.f, 0.f, 0.f, 0.f, 0.f, 0.f};
  v8f acc1 = acc0;
  wmma_mainloop(Ab, Ec, Bb0, Bb1, KEc, KEc, acc0, acc1);

  const int lane = threadIdx.x & 31;
  const int col  = lane & 15;
  const int row0 = (lane >> 4) * 8;
#pragma unroll
  for (int h = 0; h < 2; ++h) {
    const int n = n0 + h * 16 + col;
    const float bv = bias[n];
    const v8f& acc = h ? acc1 : acc0;
    v8bf st;
#pragma unroll
    for (int r = 0; r < 8; ++r) {
      float v = acc[r] + bv;
      v = v > 0.f ? v : 0.f;
      const size_t gr = (size_t)(b * Lc + l0 + row0 + r);
      out_f32[gr * ldc + n] = v;
      st[r] = (__bf16)v;
    }
    // transposed bf16 store: 8 consecutive l -> one contiguous 16B store
    *(v8bf*)(outT_bf16 + ((size_t)b * Nout + n) * Lc + l0 + row0) = st;
  }
}

// ---------------------------------------------------------------------------
// Batched GEMM, NT, 16x32 tile per wave:
//   out[b,i,j] = sum_k A[b,i,k] * Bt[b,j,k]   (+ addsrc[idx])
// f32 out indexed as outf[b*obatch + i*ldc + j]; bf16 out contiguous [b,M,N].
// ---------------------------------------------------------------------------
__global__ __launch_bounds__(32)
void gemm_nt2(const __bf16* __restrict__ A, const __bf16* __restrict__ Bt,
              int M, int N, int Kd,
              float* __restrict__ outf, int ldc, size_t obatch,
              const float* __restrict__ addsrc, __bf16* __restrict__ outb) {
  const int ntn = N / 32, ntm = M / 16;
  const int tile = blockIdx.x;
  const int b  = tile / (ntm * ntn);
  const int rm = tile % (ntm * ntn);
  const int mt = rm / ntn, nt = rm % ntn;
  const __bf16* Ab  = A  + (size_t)b * M * Kd + (size_t)(mt * 16) * Kd;
  const __bf16* Bb0 = Bt + (size_t)b * N * Kd + (size_t)(nt * 32) * Kd;
  const __bf16* Bb1 = Bb0 + (size_t)16 * Kd;

  v8f acc0 = {0.f, 0.f, 0.f, 0.f, 0.f, 0.f, 0.f, 0.f};
  v8f acc1 = acc0;
  wmma_mainloop(Ab, Kd, Bb0, Bb1, Kd, Kd, acc0, acc1);

  const int lane = threadIdx.x & 31;
  const int col  = lane & 15;
  const int row0 = (lane >> 4) * 8;
#pragma unroll
  for (int h = 0; h < 2; ++h) {
    const int n = nt * 32 + h * 16 + col;
    const v8f& acc = h ? acc1 : acc0;
#pragma unroll
    for (int r = 0; r < 8; ++r) {
      const int row = mt * 16 + row0 + r;
      float v = acc[r];
      if (outf) {
        const size_t idx = (size_t)b * obatch + (size_t)row * ldc + n;
        if (addsrc) v += addsrc[idx];
        outf[idx] = v;
      }
      if (outb) outb[(size_t)b * M * N + (size_t)row * N + n] = (__bf16)v;
    }
  }
}

// ---------------------------------------------------------------------------
// Row-wise l2 normalize (reads strided f32, writes contiguous bf16)
// ---------------------------------------------------------------------------
__global__ void l2norm_rows(const float* __restrict__ src, int ld, int ncols,
                            __bf16* __restrict__ dst, int nrows) {
  int r = blockIdx.x * blockDim.x + threadIdx.x;
  if (r >= nrows) return;
  const float* p = src + (size_t)r * ld;
  float ss = 0.f;
  for (int c = 0; c < ncols; ++c) ss += p[c] * p[c];
  const float inv = rsqrtf(fmaxf(ss, 1e-12f));
  __bf16* q = dst + (size_t)r * ncols;
  for (int c = 0; c < ncols; ++c) q[c] = (__bf16)(p[c] * inv);
}

// ---------------------------------------------------------------------------
// softmask2d row kernel: one 128-thread block per row (L=512, 4 per thread)
// ---------------------------------------------------------------------------
__global__ __launch_bounds__(128)
void softmask_rows(const float* __restrict__ X, const float* __restrict__ mask,
                   int scale, __bf16* __restrict__ outb) {
  const int row = blockIdx.x;           // b*L + i
  const int b = row / Lc, i = row % Lc;
  const float* xp = X + (size_t)row * Lc;
  const float* mr = mask + (size_t)b * Lc;
  const int t = threadIdx.x;
  float v[4];
#pragma unroll
  for (int q = 0; q < 4; ++q) v[q] = xp[t + q * 128];

  __shared__ float red[128];
  float mx = 0.f;
  if (scale) {
    mx = -3.4e38f;
#pragma unroll
    for (int q = 0; q < 4; ++q) mx = fmaxf(mx, v[q]);
    red[t] = mx; __syncthreads();
    for (int s = 64; s > 0; s >>= 1) {
      if (t < s) red[t] = fmaxf(red[t], red[t + s]);
      __syncthreads();
    }
    mx = red[0]; __syncthreads();
  }
  float s = 0.f;
#pragma unroll
  for (int q = 0; q < 4; ++q) {
    float e = __expf(v[q] - mx) * mr[t + q * 128];
    v[q] = e; s += e;
  }
  red[t] = s; __syncthreads();
  for (int st = 64; st > 0; st >>= 1) {
    if (t < st) red[t] += red[t + st];
    __syncthreads();
  }
  const float inv = 1.f / (red[0] + 1e-7f);
  const float mi  = mask[(size_t)b * Lc + i];
#pragma unroll
  for (int q = 0; q < 4; ++q)
    outb[(size_t)row * Lc + t + q * 128] = (__bf16)(v[q] * inv * mi);
}

// Final attention: softmask(scale=True) + att_ao + rank-1 opinion propagation
__global__ __launch_bounds__(128)
void softmask_final(const float* __restrict__ X, const float* __restrict__ mask,
                    const __bf16* __restrict__ attao, const float* __restrict__ oc,
                    const float* __restrict__ maskS, const float* __restrict__ colsum,
                    __bf16* __restrict__ outb) {
  const int row = blockIdx.x;
  const int b = row / Lc, i = row % Lc;
  const float* xp = X + (size_t)row * Lc;
  const float* mr = mask + (size_t)b * Lc;
  const int t = threadIdx.x;
  float v[4];
#pragma unroll
  for (int q = 0; q < 4; ++q) v[q] = xp[t + q * 128];

  __shared__ float red[128];
  float mx = -3.4e38f;
#pragma unroll
  for (int q = 0; q < 4; ++q) mx = fmaxf(mx, v[q]);
  red[t] = mx; __syncthreads();
  for (int s = 64; s > 0; s >>= 1) {
    if (t < s) red[t] = fmaxf(red[t], red[t + s]);
    __syncthreads();
  }
  mx = red[0]; __syncthreads();

  float s = 0.f;
#pragma unroll
  for (int q = 0; q < 4; ++q) {
    float e = __expf(v[q] - mx) * mr[t + q * 128];
    v[q] = e; s += e;
  }
  red[t] = s; __syncthreads();
  for (int st = 64; st > 0; st >>= 1) {
    if (t < st) red[t] += red[t + st];
    __syncthreads();
  }
  const float inv  = 1.f / (red[0] + 1e-7f);
  const float mi   = mask[(size_t)b * Lc + i];
  const float prop = oc[row] * maskS[b];
#pragma unroll
  for (int q = 0; q < 4; ++q) {
    const int j = t + q * 128;
    const size_t idx = (size_t)row * Lc + j;
    float a = v[q] * inv * mi + (float)attao[idx] + prop * colsum[(size_t)b * Lc + j];
    outb[idx] = (__bf16)a;
  }
}

// ---------------------------------------------------------------------------
// Dense heads (N=3): out[r,c] = bias[c] + sum_k X[r,k]*W[k,c]
// ---------------------------------------------------------------------------
__global__ void dense_pred(const float* __restrict__ X, int ld, int Kd,
                           const float* __restrict__ W, const float* __restrict__ bias,
                           float* __restrict__ out, int nrows) {
  int r = blockIdx.x * blockDim.x + threadIdx.x;
  if (r >= nrows) return;
  float a0 = bias[0], a1 = bias[1], a2 = bias[2];
  const float* p = X + (size_t)r * ld;
  for (int k = 0; k < Kd; ++k) {
    const float x = p[k];
    a0 += x * W[k * 3 + 0];
    a1 += x * W[k * 3 + 1];
    a2 += x * W[k * 3 + 2];
  }
  out[(size_t)r * 3 + 0] = a0;
  out[(size_t)r * 3 + 1] = a1;
  out[(size_t)r * 3 + 2] = a2;
}

__global__ void opinion_conf(const float* __restrict__ pred,
                             float* __restrict__ oc, int nrows) {
  int r = blockIdx.x * blockDim.x + threadIdx.x;
  if (r >= nrows) return;
  const float p0 = pred[(size_t)r * 3], p1 = pred[(size_t)r * 3 + 1],
              p2 = pred[(size_t)r * 3 + 2];
  const float m  = fmaxf(p0, fmaxf(p1, p2));
  const float e0 = __expf(p0 - m), e1 = __expf(p1 - m), e2 = __expf(p2 - m);
  const float sm0 = e0 / (e0 + e1 + e2);
  oc[r] = fmaxf(0.f, 1.f - 2.f * sm0);
}

__global__ void mask_rowsum(const float* __restrict__ mask, float* __restrict__ out) {
  int b = threadIdx.x;
  if (b < Bc) {
    float s = 0.f;
    for (int j = 0; j < Lc; ++j) s += mask[(size_t)b * Lc + j];
    out[b] = s;
  }
}

__global__ void pos_colsum(const float* __restrict__ pos, float* __restrict__ out) {
  int idx = blockIdx.x * blockDim.x + threadIdx.x;
  if (idx >= Bc * Lc) return;
  const int b = idx / Lc, k = idx % Lc;
  const float* p = pos + (size_t)b * Lc * Lc + k;
  float s = 0.f;
  for (int j = 0; j < Lc; ++j) s += p[(size_t)j * Lc];
  out[idx] = s;
}

// ---------------------------------------------------------------------------
// Host-side orchestration
// ---------------------------------------------------------------------------
extern "C" void kernel_launch(void* const* d_in, const int* in_sizes, int n_in,
                              void* d_out, int out_size, void* d_ws, size_t ws_size,
                              hipStream_t stream) {
  (void)in_sizes; (void)n_in; (void)out_size; (void)ws_size;

  const float* aspect_in  = (const float*)d_in[0];
  const float* opinion_in = (const float*)d_in[1];
  const float* context_in = (const float*)d_in[2];
  const float* query_in   = (const float*)d_in[3];
  const float* word_mask  = (const float*)d_in[4];
  const float* pos_att    = (const float*)d_in[5];
  const float* Wa = (const float*)d_in[6];  const float* ba = (const float*)d_in[7];
  const float* Wo = (const float*)d_in[8];  const float* bo = (const float*)d_in[9];
  const float* Wc = (const float*)d_in[10]; const float* bc = (const float*)d_in[11];
  const float* Wda = (const float*)d_in[12]; const float* bda = (const float*)d_in[13];
  const float* Wdo = (const float*)d_in[14]; const float* bdo = (const float*)d_in[15];
  const float* Wds = (const float*)d_in[16]; const float* bds = (const float*)d_in[17];

  char*  ws  = (char*)d_ws;
  float* out = (float*)d_out;

  __bf16* xa_pad   = (__bf16*)(ws + OFF_XAP);
  __bf16* xo_pad   = (__bf16*)(ws + OFF_XOP);
  __bf16* xc_pad   = (__bf16*)(ws + OFF_XCP);
  __bf16* q_bf     = (__bf16*)(ws + OFF_Q);
  __bf16* posT_bf  = (__bf16*)(ws + OFF_POST);
  __bf16* WaT      = (__bf16*)(ws + OFF_WAT);
  __bf16* WoT      = (__bf16*)(ws + OFF_WOT);
  __bf16* WcT      = (__bf16*)(ws + OFF_WCT);
  __bf16* aconvT   = (__bf16*)(ws + OFF_ACONVT);
  __bf16* oconvT   = (__bf16*)(ws + OFF_OCONVT);
  __bf16* cconvT   = (__bf16*)(ws + OFF_CCONVT);
  __bf16* an_bf    = (__bf16*)(ws + OFF_AN);
  __bf16* on_bf    = (__bf16*)(ws + OFF_ON);
  __bf16* cn_bf    = (__bf16*)(ws + OFF_CN);
  __bf16* attao_bf = (__bf16*)(ws + OFF_ATTAO);
  __bf16* attoa_bf = (__bf16*)(ws + OFF_ATTOA);
  __bf16* tmp1_bf  = (__bf16*)(ws + OFF_TMP1);
  __bf16* fatt_bf  = (__bf16*)(ws + OFF_FATT);
  float*  score_f  = (float*)(ws + OFF_SCORE);
  float*  oc_f     = (float*)(ws + OFF_OC);
  float*  maskS_f  = (float*)(ws + OFF_MS);
  float*  colsum_f = (float*)(ws + OFF_CS);

  const int nrows = Bc * Lc;  // 16384

  // --- precision conversion + padding + transposes ---
  cvt_pad_f32_bf16<<<(unsigned)((N_BLEP + 255) / 256), 256, 0, stream>>>(aspect_in,  xa_pad);
  cvt_pad_f32_bf16<<<(unsigned)((N_BLEP + 255) / 256), 256, 0, stream>>>(opinion_in, xo_pad);
  cvt_pad_f32_bf16<<<(unsigned)((N_BLEP + 255) / 256), 256, 0, stream>>>(context_in, xc_pad);
  cvt_f32_bf16<<<(unsigned)((N_BLE + 255) / 256), 256, 0, stream>>>(query_in, q_bf, N_BLE);
  transpose_f32_bf16<<<(unsigned)((N_BLL + 255) / 256), 256, 0, stream>>>(pos_att, posT_bf);
  wt_transpose<<<(unsigned)((N_WAF + 255) / 256), 256, 0, stream>>>(Wa, WaT, KEc, Fc);
  wt_transpose<<<(unsigned)((N_WAF + 255) / 256), 256, 0, stream>>>(Wo, WoT, KEc, Fc);
  wt_transpose<<<(unsigned)((N_WCE + 255) / 256), 256, 0, stream>>>(Wc, WcT, KEc, Ec);

  // --- conv1d + relu (WMMA on padded input), bf16 copies transposed [b,N,L] ---
  conv1d_wmma<<<Bc * 32 * (Fc / 32), 32, 0, stream>>>(xa_pad, WaT, ba,
      out + DO_AINT, 1024, aconvT, Fc);
  conv1d_wmma<<<Bc * 32 * (Fc / 32), 32, 0, stream>>>(xo_pad, WoT, bo,
      out + DO_OINT, 1024, oconvT, Fc);
  conv1d_wmma<<<Bc * 32 * (Ec / 32), 32, 0, stream>>>(xc_pad, WcT, bc,
      out + DO_CCONV, Ec, cconvT, Ec);

  // --- l2 normalize ---
  l2norm_rows<<<(nrows + 255) / 256, 256, 0, stream>>>(out + DO_AINT, 1024, Fc, an_bf, nrows);
  l2norm_rows<<<(nrows + 255) / 256, 256, 0, stream>>>(out + DO_OINT, 1024, Fc, on_bf, nrows);
  l2norm_rows<<<(nrows + 255) / 256, 256, 0, stream>>>(out + DO_CCONV, Ec,  Ec, cn_bf, nrows);

  // --- bidirectional masked attention (all-NT WMMA) ---
  gemm_nt2<<<Bc * 32 * (Lc / 32), 32, 0, stream>>>(an_bf, on_bf, Lc, Lc, Fc,
      score_f, Lc, (size_t)Lc * Lc, (const float*)nullptr, (__bf16*)nullptr);
  softmask_rows<<<nrows, 128, 0, stream>>>(score_f, word_mask, 0, attao_bf);
  gemm_nt2<<<Bc * 32 * (Lc / 32), 32, 0, stream>>>(on_bf, an_bf, Lc, Lc, Fc,
      score_f, Lc, (size_t)Lc * Lc, (const float*)nullptr, (__bf16*)nullptr);
  softmask_rows<<<nrows, 128, 0, stream>>>(score_f, word_mask, 0, attoa_bf);

  // att @ conv: B^T is the transposed conv copy -> NT fast path
  gemm_nt2<<<Bc * 32 * (Fc / 32), 32, 0, stream>>>(attao_bf, oconvT, Lc, Fc, Lc,
      out + DO_AINT + 512, 1024, (size_t)Lc * 1024, (const float*)nullptr, (__bf16*)nullptr);
  gemm_nt2<<<Bc * 32 * (Fc / 32), 32, 0, stream>>>(attoa_bf, aconvT, Lc, Fc, Lc,
      out + DO_OINT + 512, 1024, (size_t)Lc * 1024, (const float*)nullptr, (__bf16*)nullptr);

  // --- dense heads for aspect/opinion ---
  dense_pred<<<(nrows + 255) / 256, 256, 0, stream>>>(out + DO_AINT, 1024, 1024, Wda, bda, out + DO_APRED, nrows);
  dense_pred<<<(nrows + 255) / 256, 256, 0, stream>>>(out + DO_OINT, 1024, 1024, Wdo, bdo, out + DO_OPRED, nrows);

  // --- opinion propagation (rank-1 factorization) ---
  opinion_conf<<<(nrows + 255) / 256, 256, 0, stream>>>(out + DO_OPRED, oc_f, nrows);
  mask_rowsum<<<1, 32, 0, stream>>>(word_mask, maskS_f);
  pos_colsum<<<(nrows + 255) / 256, 256, 0, stream>>>(pos_att, colsum_f);

  // --- context attention ---
  gemm_nt2<<<Bc * 32 * (Lc / 32), 32, 0, stream>>>(q_bf, cn_bf, Lc, Lc, Ec,
      (float*)nullptr, 0, 0, (const float*)nullptr, tmp1_bf);
  gemm_nt2<<<Bc * 32 * (Lc / 32), 32, 0, stream>>>(tmp1_bf, posT_bf, Lc, Lc, Lc,
      score_f, Lc, (size_t)Lc * Lc, (const float*)nullptr, (__bf16*)nullptr);
  softmask_final<<<nrows, 128, 0, stream>>>(score_f, word_mask, attao_bf, oc_f,
      maskS_f, colsum_f, fatt_bf);
  gemm_nt2<<<Bc * 32 * (Ec / 32), 32, 0, stream>>>(fatt_bf, cconvT, Lc, Ec, Lc,
      out + DO_CINT, Ec, (size_t)Lc * Ec, query_in, (__bf16*)nullptr);

  // --- sentiment head ---
  dense_pred<<<(nrows + 255) / 256, 256, 0, stream>>>(out + DO_CINT, Ec, Ec, Wds, bds, out + DO_SPRED, nrows);
}
